// AnatomicalGCN_30511447671338
// MI455X (gfx1250) — matrix operand order repaired and checked by reading, not compiled, not run on hardware
//
#include <hip/hip_runtime.h>
#include <hip/hip_bf16.h>
#include <cstdint>

// ---------------------------------------------------------------------------
// MI455X (gfx1250) implementation, round 3.
//  - wave32; WMMA f32_16x16x32_f16 for all dense GEMM stages
//  - async global->LDS (ASYNCcnt) staging for f16 weight tiles
//  - f16 pre-converted weights; float4 / f16x4 vectorized LDS staging
// ---------------------------------------------------------------------------

typedef _Float16 f16;
typedef __attribute__((ext_vector_type(16))) _Float16 v16h;
typedef __attribute__((ext_vector_type(4)))  _Float16 h4;
typedef __attribute__((ext_vector_type(8)))  float    v8f;

#define BN_RSQ 0.9999950000375f   // 1/sqrt(1+1e-5)

__device__ __forceinline__ float wred_sum(float v){
  #pragma unroll
  for (int o = 16; o > 0; o >>= 1) v += __shfl_xor(v, o, 32);
  return v;
}
__device__ __forceinline__ float wred_max(float v){
  #pragma unroll
  for (int o = 16; o > 0; o >>= 1) v = fmaxf(v, __shfl_xor(v, o, 32));
  return v;
}
__device__ __forceinline__ h4 cvt4(float4 f){
  h4 r; r[0]=(f16)f.x; r[1]=(f16)f.y; r[2]=(f16)f.z; r[3]=(f16)f.w; return r;
}
// Async 16B global->LDS copy (GLOBAL_LOAD_ASYNC_TO_LDS_B128, ASYNCcnt)
__device__ __forceinline__ void async_cp16(void* lds, const void* g){
  unsigned l = (unsigned)(uintptr_t)lds;
  unsigned long long ga = (unsigned long long)(uintptr_t)g;
  asm volatile("global_load_async_to_lds_b128 %0, %1, off"
               :: "v"(l), "v"(ga) : "memory");
}
__device__ __forceinline__ void async_wait0(){
  asm volatile("s_wait_asynccnt 0x0" ::: "memory");
}

// CDNA5 WMMA f16 fragment index helpers (16x16x32, wave32)
__device__ __forceinline__ int a_kidx(int lane, int v, int h){
  int base = (lane < 16) ? 0 : 8;
  return (v < 4) ? (base + 2*v + h) : (base + 16 + 2*(v-4) + h);
}
__device__ __forceinline__ int b_kidx(int lane, int v, int h){
  int base = (lane < 16) ? 0 : 8;
  return (v < 4) ? (base + 2*v + h) : (base + 16 + 2*(v-4) + h);
}

// ===========================================================================
// Weight convert: f32 (N,K) -> f16 (N,Kpad), zero padded
// ===========================================================================
__global__ void w2h_kernel(const float* __restrict__ W, f16* __restrict__ Wh,
                           int K, int Kpad, size_t total)
{
  size_t i = (size_t)blockIdx.x * 256 + threadIdx.x;
  if (i >= total) return;
  size_t n = i / (size_t)Kpad; int k = (int)(i % (size_t)Kpad);
  Wh[i] = (k < K) ? (f16)W[n*(size_t)K + k] : (f16)0.f;
}

// ===========================================================================
// Region encoder (one wave per sample, lane == channel d)
// ===========================================================================
template<int N>
__global__ __launch_bounds__(256) void region_kernel(
    const float* __restrict__ x,
    const float* __restrict__ W1, const float* __restrict__ b1,
    const float* __restrict__ W2, const float* __restrict__ b2,
    const float* __restrict__ lng, const float* __restrict__ lnb,
    const float* __restrict__ Wqkv, const float* __restrict__ bqkv,
    const float* __restrict__ Wo,  const float* __restrict__ bo,
    const float* __restrict__ ilng, const float* __restrict__ ilnb,
    float* __restrict__ R, int region)
{
  __shared__ float sW1[320], sb1[32], sW2[1024], sb2[32];
  __shared__ float slng[64], slnb[64], silg[64], silb[64];
  __shared__ float sx[8][200];
  __shared__ float sh1[8][640];
  __shared__ __align__(16) float sp[8][64];
  __shared__ __align__(16) float sv[8][64];

  int tid = threadIdx.x, lane = tid & 31, w = tid >> 5;
  for (int i = tid; i < 320;  i += 256) sW1[i] = W1[i];
  for (int i = tid; i < 1024; i += 256) sW2[i] = W2[i];
  if (tid < 32) { sb1[tid] = b1[tid]; sb2[tid] = b2[tid]; }
  if (tid < 64) { slng[tid] = lng[tid]; slnb[tid] = lnb[tid];
                  silg[tid] = ilng[tid]; silb[tid] = ilnb[tid]; }

  size_t s = (size_t)blockIdx.x * 8 + w;
  for (int i = lane; i < N*10; i += 32) sx[w][i] = x[s*(size_t)(N*10) + i];
  __syncthreads();

  float dinv[N], dself[N];
  #pragma unroll
  for (int n = 0; n < N; n++){
    int cnt = (n>=1) + (n+1<N) + (n>=2) + (n+2<N);
    float deg = (float)(cnt + 1);
    dinv[n] = rsqrtf(deg);
    dself[n] = 1.f/deg;
  }
  int d = lane;
  float xw[N];
  #pragma unroll
  for (int n = 0; n < N; n++){
    float a = 0.f;
    #pragma unroll
    for (int c = 0; c < 10; c++) a += sx[w][n*10+c] * sW1[c*32+d];
    xw[n] = a;
  }
  #pragma unroll
  for (int n = 0; n < N; n++){
    float nb = 0.f;
    if (n>=1)  nb += xw[n-1]*dinv[n-1];
    if (n+1<N) nb += xw[n+1]*dinv[n+1];
    if (n>=2)  nb += xw[n-2]*dinv[n-2];
    if (n+2<N) nb += xw[n+2]*dinv[n+2];
    sh1[w][n*32+d] = fmaxf(nb*dinv[n] + xw[n]*dself[n] + sb1[d], 0.f);
  }
  __syncthreads();
  #pragma unroll
  for (int n = 0; n < N; n++){
    float a = 0.f;
    #pragma unroll
    for (int c = 0; c < 32; c++) a += sh1[w][n*32+c] * sW2[c*32+d];
    xw[n] = a;
  }
  float pm = 0.f, px = -3.4e38f;
  #pragma unroll
  for (int n = 0; n < N; n++){
    float nb = 0.f;
    if (n>=1)  nb += xw[n-1]*dinv[n-1];
    if (n+1<N) nb += xw[n+1]*dinv[n+1];
    if (n>=2)  nb += xw[n-2]*dinv[n-2];
    if (n+2<N) nb += xw[n+2]*dinv[n+2];
    float a = fmaxf(nb*dinv[n] + xw[n]*dself[n] + sb2[d], 0.f);
    pm += a; px = fmaxf(px, a);
  }
  pm /= (float)N;
  float mean = wred_sum(pm + px) / 64.f;
  float t0 = pm - mean, t1 = px - mean;
  float var = wred_sum(t0*t0 + t1*t1) / 64.f;
  float rstd = rsqrtf(var + 1e-5f);
  float p0 = t0*rstd*slng[d]    + slnb[d];
  float p1 = t1*rstd*slng[32+d] + slnb[32+d];
  sp[w][d] = p0; sp[w][32+d] = p1;
  __syncthreads();
  // v = Wv * p + bv  (rows 128..191 of intra Wqkv), float4 dot
  float v0 = bqkv[128+d], v1 = bqkv[160+d];
  {
    const float4* w0 = (const float4*)&Wqkv[(size_t)(128+d)*64];
    const float4* w1v = (const float4*)&Wqkv[(size_t)(160+d)*64];
    const float4* pp = (const float4*)&sp[w][0];
    #pragma unroll
    for (int c = 0; c < 16; c++){
      float4 p4 = pp[c], a4 = w0[c], b4 = w1v[c];
      v0 += p4.x*a4.x + p4.y*a4.y + p4.z*a4.z + p4.w*a4.w;
      v1 += p4.x*b4.x + p4.y*b4.y + p4.z*b4.z + p4.w*b4.w;
    }
  }
  sv[w][d] = v0; sv[w][32+d] = v1;
  __syncthreads();
  float a0 = bo[d], a1 = bo[32+d];
  {
    const float4* w0 = (const float4*)&Wo[(size_t)d*64];
    const float4* w1v = (const float4*)&Wo[(size_t)(32+d)*64];
    const float4* vv = (const float4*)&sv[w][0];
    #pragma unroll
    for (int c = 0; c < 16; c++){
      float4 p4 = vv[c], a4 = w0[c], b4 = w1v[c];
      a0 += p4.x*a4.x + p4.y*a4.y + p4.z*a4.z + p4.w*a4.w;
      a1 += p4.x*b4.x + p4.y*b4.y + p4.z*b4.z + p4.w*b4.w;
    }
  }
  float r0 = p0 + a0, r1 = p1 + a1;
  mean = wred_sum(r0 + r1) / 64.f;
  t0 = r0 - mean; t1 = r1 - mean;
  var = wred_sum(t0*t0 + t1*t1) / 64.f;
  rstd = rsqrtf(var + 1e-5f);
  R[(s*6 + region)*64 + d]      = t0*rstd*silg[d]    + silb[d];
  R[(s*6 + region)*64 + 32 + d] = t1*rstd*silg[32+d] + silb[32+d];
}

// ===========================================================================
// Pack concat(all regions) -> (BT, 704), cols 690..703 zero
// ===========================================================================
__global__ void pack_kernel(const float* __restrict__ l, const float* __restrict__ r,
                            const float* __restrict__ le, const float* __restrict__ re,
                            const float* __restrict__ no, const float* __restrict__ mo,
                            float* __restrict__ P, size_t total)
{
  size_t i = (size_t)blockIdx.x * 256 + threadIdx.x;
  if (i >= total) return;
  size_t bt = i / 704; int k = (int)(i % 704);
  float v;
  if      (k < 90)  v = l [bt*90  + k];
  else if (k < 180) v = r [bt*90  + (k-90)];
  else if (k < 290) v = le[bt*110 + (k-180)];
  else if (k < 400) v = re[bt*110 + (k-290)];
  else if (k < 490) v = no[bt*90  + (k-400)];
  else if (k < 690) v = mo[bt*200 + (k-490)];
  else              v = 0.f;
  P[i] = v;
}

// ===========================================================================
// WMMA GEMM: C(MxN) = A(MxK_f32) @ Wh(N,K_f16)^T + bias ; act 0/1/2(BN+relu)
// Requires M%64==0, N%32==0, K%32==0, lda%4==0.  Block 256 = 8 waves, 64x32.
// B-tile staged with async global->LDS (no VGPR round trip).
// ===========================================================================
__global__ __launch_bounds__(256) void gemm_wmma(
    const float* __restrict__ A, int lda,
    const f16* __restrict__ Wh,
    const float* __restrict__ bias,
    const float* __restrict__ bng, const float* __restrict__ bnb,
    float* __restrict__ C, int ldc,
    int N, int K, int act)
{
  __shared__ __align__(16) f16 As[64][36];
  __shared__ __align__(16) f16 Ws[32][40];   // 80B rows: 16B-aligned async stores
  int tid = threadIdx.x, lane = tid & 31, w = tid >> 5;
  int wm = w >> 1, wn = w & 1;
  int m0 = blockIdx.y * 64, n0 = blockIdx.x * 32;
  v8f acc = {};
  for (int k0 = 0; k0 < K; k0 += 32){
    // B tile: 32x32 f16 = 2048B, 128 async 16B copies
    if (tid < 128){
      int n = tid >> 2, kq = (tid & 3) * 8;
      async_cp16(&Ws[n][kq], &Wh[(size_t)(n0+n)*K + k0 + kq]);
    }
    // A tile: 64x32 f32 -> f16, 512 float4 loads, 8B LDS stores
    #pragma unroll
    for (int it = 0; it < 2; it++){
      int e = tid + it*256;
      int m = e >> 3, kq = (e & 7) * 4;
      float4 f = *(const float4*)&A[(size_t)(m0+m)*lda + k0 + kq];
      *(h4*)&As[m][kq] = cvt4(f);
    }
    if (k0 + 32 < K)
      __builtin_prefetch(&A[(size_t)(m0 + (tid >> 2))*lda + k0 + 32], 0, 1);
    async_wait0();
    __syncthreads();
    v16h a, b;
    int mrow = wm*16 + (lane & 15);
    int ncol = wn*16 + (lane & 15);
    #pragma unroll
    for (int e = 0; e < 16; e++){
      int v = e >> 1, h = e & 1;
      a[e] = As[mrow][a_kidx(lane, v, h)];
      b[e] = Ws[ncol][b_kidx(lane, v, h)];
    }
    acc = __builtin_amdgcn_wmma_f32_16x16x32_f16(false, a, false, b, (short)0, acc, false, false);
    __syncthreads();
  }
  int n = n0 + wn*16 + (lane & 15);
  float bv = bias ? bias[n] : 0.f;
  float sg = 1.f, sb = 0.f;
  if (act == 2){ sg = bng[n] * BN_RSQ; sb = bnb[n]; }
  #pragma unroll
  for (int v = 0; v < 8; v++){
    int m = m0 + wm*16 + v + ((lane < 16) ? 0 : 8);
    float y = acc[v] + bv;
    if (act == 2) y = y*sg + sb;
    if (act >= 1) y = fmaxf(y, 0.f);
    C[(size_t)m*ldc + n] = y;
  }
}

// ===========================================================================
// Cross attention over 6 region tokens (nh=4, hd=16); zeroed slot keeps biases.
// ===========================================================================
__global__ __launch_bounds__(256) void cross_kernel(
    const float* __restrict__ R, const float* __restrict__ Wqkv,
    const float* __restrict__ bqkv, const float* __restrict__ Wo,
    const float* __restrict__ bo, const float* __restrict__ lng,
    const float* __restrict__ lnb, float* __restrict__ CR)
{
  __shared__ float sb[192];
  __shared__ __align__(16) float sR[8][384];
  __shared__ __align__(16) float sP[8][1152];
  __shared__ float ssc[8][24];
  __shared__ float swt[8][24];
  __shared__ __align__(16) float sout[8][64];
  int tid = threadIdx.x, lane = tid & 31, w = tid >> 5;
  if (tid < 192) sb[tid] = bqkv[tid];
  size_t s = (size_t)blockIdx.x * 8 + w;
  for (int i = lane; i < 96; i += 32)
    *(float4*)&sR[w][i*4] = *(const float4*)&R[s*384 + i*4];
  __syncthreads();
  for (int idx = lane; idx < 1152; idx += 32){
    int j = idx / 192, o = idx % 192;
    float a = sb[o];
    const float4* wr = (const float4*)&Wqkv[(size_t)o*64];
    const float4* rr = (const float4*)&sR[w][j*64];
    #pragma unroll
    for (int c = 0; c < 16; c++){
      float4 x4 = rr[c], w4 = wr[c];
      a += x4.x*w4.x + x4.y*w4.y + x4.z*w4.z + x4.w*w4.w;
    }
    sP[w][idx] = a;
  }
  __syncthreads();
  int d = lane;
  float acc0 = 0.f, acc1 = 0.f;
  for (int i = 0; i < 6; i++){
    if (lane < 24){
      int j = lane >> 2, h = lane & 3;
      float a = 0.f;
      for (int dd = 0; dd < 16; dd++){
        float q = sP[w][i*192 + h*16 + dd];
        float k = (j == i) ? sb[64 + h*16 + dd] : sP[w][j*192 + 64 + h*16 + dd];
        a += q*k;
      }
      ssc[w][lane] = a * 0.25f;
    }
    __syncthreads();
    if (lane < 4){
      int h = lane;
      float m = -3.4e38f;
      for (int j = 0; j < 6; j++) m = fmaxf(m, ssc[w][j*4+h]);
      float e[6], sum = 0.f;
      for (int j = 0; j < 6; j++){ e[j] = __expf(ssc[w][j*4+h] - m); sum += e[j]; }
      for (int j = 0; j < 6; j++) swt[w][j*4+h] = e[j]/sum;
    }
    __syncthreads();
    for (int o = lane; o < 64; o += 32){
      int h = o >> 4;
      float a = 0.f;
      for (int j = 0; j < 6; j++){
        float v = (j == i) ? sb[128+o] : sP[w][j*192 + 128 + o];
        a += swt[w][j*4+h] * v;
      }
      sout[w][o] = a;
    }
    __syncthreads();
    float a0 = bo[d], a1 = bo[32+d];
    {
      const float4* w0 = (const float4*)&Wo[(size_t)d*64];
      const float4* w1v = (const float4*)&Wo[(size_t)(32+d)*64];
      const float4* oo = (const float4*)&sout[w][0];
      #pragma unroll
      for (int c = 0; c < 16; c++){
        float4 p4 = oo[c], a4 = w0[c], b4 = w1v[c];
        a0 += p4.x*a4.x + p4.y*a4.y + p4.z*a4.z + p4.w*a4.w;
        a1 += p4.x*b4.x + p4.y*b4.y + p4.z*b4.z + p4.w*b4.w;
      }
    }
    float r0 = sR[w][i*64 + d] + a0, r1 = sR[w][i*64 + 32 + d] + a1;
    float mean = wred_sum(r0 + r1) / 64.f;
    float t0 = r0 - mean, t1 = r1 - mean;
    float var = wred_sum(t0*t0 + t1*t1) / 64.f;
    float rstd = rsqrtf(var + 1e-5f);
    acc0 += t0*rstd*lng[d]    + lnb[d];
    acc1 += t1*rstd*lng[32+d] + lnb[32+d];
    __syncthreads();
  }
  CR[s*64 + d]      = acc0 * (1.f/6.f);
  CR[s*64 + 32 + d] = acc1 * (1.f/6.f);
}

// ===========================================================================
// per_frame = LN(concat(cross, g)) + posenc(t) ; lane holds 4 contiguous dims
// ===========================================================================
__global__ __launch_bounds__(256) void perframe_kernel(
    const float* __restrict__ CR, const float* __restrict__ G,
    const float* __restrict__ g, const float* __restrict__ bb,
    float* __restrict__ PF)
{
  int tid = threadIdx.x, lane = tid & 31, w = tid >> 5;
  size_t s = (size_t)blockIdx.x * 8 + w;
  int t = (int)(s & 511);
  int d0 = lane * 4;
  float4 x4 = (lane < 16) ? *(const float4*)&CR[s*64 + d0]
                          : *(const float4*)&G[s*64 + d0 - 64];
  float v[4] = {x4.x, x4.y, x4.z, x4.w};
  float sum = v[0]+v[1]+v[2]+v[3];
  sum = wred_sum(sum); float mean = sum / 128.f;
  float var = 0.f;
  #pragma unroll
  for (int i = 0; i < 4; i++){ float q = v[i]-mean; var += q*q; }
  var = wred_sum(var) / 128.f;
  float rstd = rsqrtf(var + 1e-5f);
  float4 g4 = *(const float4*)&g[d0];
  float4 b4 = *(const float4*)&bb[d0];
  float gg[4] = {g4.x,g4.y,g4.z,g4.w}, bbv[4] = {b4.x,b4.y,b4.z,b4.w};
  float out[4];
  #pragma unroll
  for (int i = 0; i < 4; i++){
    int dd = d0 + i;
    int pi = dd >> 1;
    float div = __expf(-(float)(2*pi) * (9.210340371976184f / 128.f));
    float pe = (dd & 1) ? __cosf((float)t * div) : __sinf((float)t * div);
    out[i] = (v[i]-mean)*rstd*gg[i] + bbv[i] + pe;
  }
  *(float4*)&PF[s*128 + d0] = make_float4(out[0], out[1], out[2], out[3]);
}

// ===========================================================================
// Temporal attention, one wave per (b, head, 16-query tile); hd=32, T=512
// ===========================================================================
__global__ __launch_bounds__(32) void attn_kernel(
    const float* __restrict__ QKV, const int* __restrict__ lens,
    float* __restrict__ AO)
{
  __shared__ __align__(16) f16 Qs[16][36];
  __shared__ __align__(16) f16 KVs[32][36];
  __shared__ __align__(16) float Sc[16][512];
  __shared__ __align__(16) f16 Pp[16][512];
  int lane = threadIdx.x;
  int q0 = blockIdx.x * 16, hid = blockIdx.y, b = blockIdx.z;
  int len = lens[b];
  const float scale = 0.17677669529663687f;
  size_t rowb = (size_t)b * 512;
  #pragma unroll
  for (int it = 0; it < 4; it++){
    int e = lane + it*32;            // 128 float4 = 16x32
    int m = e >> 3, kq = (e & 7) * 4;
    float4 f = *(const float4*)&QKV[(rowb + q0 + m)*384 + hid*32 + kq];
    *(h4*)&Qs[m][kq] = cvt4(f);
  }
  __syncthreads();
  v16h qa;
  #pragma unroll
  for (int e = 0; e < 16; e++){
    int v = e >> 1, h = e & 1;
    qa[e] = Qs[lane & 15][a_kidx(lane, v, h)];
  }
  for (int kt = 0; kt < 32; kt++){
    #pragma unroll
    for (int it = 0; it < 4; it++){
      int e = lane + it*32;
      int n = e >> 3, kq = (e & 7) * 4;
      float4 f = *(const float4*)&QKV[(rowb + kt*16 + n)*384 + 128 + hid*32 + kq];
      *(h4*)&KVs[n][kq] = cvt4(f);
    }
    __syncthreads();
    v16h kb;
    #pragma unroll
    for (int e = 0; e < 16; e++){
      int v = e >> 1, h = e & 1;
      kb[e] = KVs[lane & 15][b_kidx(lane, v, h)];
    }
    v8f s = {};
    s = __builtin_amdgcn_wmma_f32_16x16x32_f16(false, qa, false, kb, (short)0, s, false, false);
    int key = kt*16 + (lane & 15);
    #pragma unroll
    for (int v = 0; v < 8; v++){
      int m = v + ((lane < 16) ? 0 : 8);
      Sc[m][key] = (key < len) ? s[v]*scale : -1e9f;
    }
    __syncthreads();
  }
  {
    int row = lane & 15, hb = (lane >> 4) * 256;
    float mx = -3.4e38f;
    for (int c = 0; c < 256; c++) mx = fmaxf(mx, Sc[row][hb + c]);
    mx = fmaxf(mx, __shfl_xor(mx, 16, 32));
    float sum = 0.f;
    for (int c = 0; c < 256; c++){
      float e = __expf(Sc[row][hb + c] - mx);
      Sc[row][hb + c] = e; sum += e;
    }
    sum += __shfl_xor(sum, 16, 32);
    float inv = 1.f / sum;
    for (int c = 0; c < 64; c++){
      float4 f = *(const float4*)&Sc[row][hb + c*4];
      f.x *= inv; f.y *= inv; f.z *= inv; f.w *= inv;
      *(h4*)&Pp[row][hb + c*4] = cvt4(f);
    }
  }
  __syncthreads();
  v8f o0 = {}, o1 = {};
  for (int kt = 0; kt < 16; kt++){
    #pragma unroll
    for (int it = 0; it < 8; it++){
      int e = lane + it*32;            // 256 float4 = 32x32
      int kk = e >> 3, kq = (e & 7) * 4;
      float4 f = *(const float4*)&QKV[(rowb + kt*32 + kk)*384 + 256 + hid*32 + kq];
      *(h4*)&KVs[kk][kq] = cvt4(f);
    }
    __syncthreads();
    v16h pa, vb0, vb1;
    #pragma unroll
    for (int e = 0; e < 16; e++){
      int v = e >> 1, h = e & 1;
      pa[e]  = Pp[lane & 15][kt*32 + a_kidx(lane, v, h)];
      int kk = b_kidx(lane, v, h);
      vb0[e] = KVs[kk][lane & 15];
      vb1[e] = KVs[kk][16 + (lane & 15)];
    }
    o0 = __builtin_amdgcn_wmma_f32_16x16x32_f16(false, pa, false, vb0, (short)0, o0, false, false);
    o1 = __builtin_amdgcn_wmma_f32_16x16x32_f16(false, pa, false, vb1, (short)0, o1, false, false);
    __syncthreads();
  }
  int n = lane & 15;
  #pragma unroll
  for (int v = 0; v < 8; v++){
    int m = v + ((lane < 16) ? 0 : 8);
    AO[(rowb + q0 + m)*128 + hid*32 + n]      = o0[v];
    AO[(rowb + q0 + m)*128 + hid*32 + 16 + n] = o1[v];
  }
}

// ===========================================================================
// Y = LN(A + B) * g + b  over 128 dims (lane holds 4 contiguous dims)
// ===========================================================================
__global__ __launch_bounds__(256) void ln_add128(
    const float* __restrict__ A, const float* __restrict__ Badd,
    const float* __restrict__ g, const float* __restrict__ bb,
    float* __restrict__ Y)
{
  int tid = threadIdx.x, lane = tid & 31, w = tid >> 5;
  size_t s = (size_t)blockIdx.x * 8 + w;
  int d0 = lane * 4;
  float4 a4 = *(const float4*)&A[s*128 + d0];
  float4 r4 = *(const float4*)&Badd[s*128 + d0];
  float v[4] = {a4.x+r4.x, a4.y+r4.y, a4.z+r4.z, a4.w+r4.w};
  float sum = wred_sum(v[0]+v[1]+v[2]+v[3]);
  float mean = sum / 128.f;
  float var = 0.f;
  #pragma unroll
  for (int i = 0; i < 4; i++){ float q = v[i]-mean; var += q*q; }
  var = wred_sum(var) / 128.f;
  float rstd = rsqrtf(var + 1e-5f);
  float4 g4 = *(const float4*)&g[d0];
  float4 b4 = *(const float4*)&bb[d0];
  float4 o;
  o.x = (v[0]-mean)*rstd*g4.x + b4.x;
  o.y = (v[1]-mean)*rstd*g4.y + b4.y;
  o.z = (v[2]-mean)*rstd*g4.z + b4.z;
  o.w = (v[3]-mean)*rstd*g4.w + b4.w;
  *(float4*)&Y[s*128 + d0] = o;
}

// ===========================================================================
// Fused FFN (128 -> 2048 -> 128), f16 weights async-staged to LDS
// ===========================================================================
__global__ __launch_bounds__(256) void ff_kernel(
    const float* __restrict__ X, const f16* __restrict__ W1h,
    const float* __restrict__ b1, const f16* __restrict__ W2h,
    const float* __restrict__ b2, float* __restrict__ OUT)
{
  __shared__ __align__(16) f16 Xs[64][132];
  __shared__ __align__(16) f16 W1s[32][136];   // 272B rows: 16B-aligned async
  __shared__ __align__(16) f16 Hs[64][36];
  __shared__ __align__(16) f16 W2s[128][40];   // 80B rows: 16B-aligned async
  int tid = threadIdx.x, lane = tid & 31, w = tid >> 5;
  int wm = w >> 1, wn = w & 1;
  int m0 = blockIdx.x * 64;
  #pragma unroll
  for (int it = 0; it < 8; it++){
    int e = tid + it*256;            // 2048 float4 = 64x128
    int m = e >> 5, kq = (e & 31) * 4;
    float4 f = *(const float4*)&X[(size_t)(m0+m)*128 + kq];
    *(h4*)&Xs[m][kq] = cvt4(f);
  }
  v8f acc[4] = {};
  for (int hc = 0; hc < 64; hc++){
    int hbase = hc * 32;
    // W1 chunk: 32x128 f16 = 8192B, 512 async 16B copies
    #pragma unroll
    for (int it = 0; it < 2; it++){
      int e = tid + it*256;
      int r = e >> 4, kq = (e & 15) * 8;
      async_cp16(&W1s[r][kq], &W1h[(size_t)(hbase+r)*128 + kq]);
    }
    // W2 chunk: 128x32 f16 = 8192B, 512 async 16B copies
    #pragma unroll
    for (int it = 0; it < 2; it++){
      int e = tid + it*256;
      int o = e >> 2, kq = (e & 3) * 8;
      async_cp16(&W2s[o][kq], &W2h[(size_t)o*2048 + hbase + kq]);
    }
    async_wait0();
    __syncthreads();
    v8f ch = {};
    #pragma unroll
    for (int ks = 0; ks < 4; ks++){
      v16h a, bb;
      #pragma unroll
      for (int e = 0; e < 16; e++){
        int v = e >> 1, h = e & 1;
        a[e]  = Xs [wm*16 + (lane & 15)][ks*32 + a_kidx(lane, v, h)];
        bb[e] = W1s[wn*16 + (lane & 15)][ks*32 + b_kidx(lane, v, h)];
      }
      ch = __builtin_amdgcn_wmma_f32_16x16x32_f16(false, a, false, bb, (short)0, ch, false, false);
    }
    int ncol = wn*16 + (lane & 15);
    float bsc = b1[hbase + ncol];
    #pragma unroll
    for (int v = 0; v < 8; v++){
      int mr = wm*16 + v + ((lane < 16) ? 0 : 8);
      Hs[mr][ncol] = (f16)fmaxf(ch[v] + bsc, 0.f);
    }
    __syncthreads();
    v16h ha;
    #pragma unroll
    for (int e = 0; e < 16; e++){
      int v = e >> 1, h = e & 1;
      ha[e] = Hs[wm*16 + (lane & 15)][a_kidx(lane, v, h)];
    }
    #pragma unroll
    for (int t = 0; t < 4; t++){
      v16h bb;
      #pragma unroll
      for (int e = 0; e < 16; e++){
        int v = e >> 1, h = e & 1;
        bb[e] = W2s[wn*64 + t*16 + (lane & 15)][b_kidx(lane, v, h)];
      }
      acc[t] = __builtin_amdgcn_wmma_f32_16x16x32_f16(false, ha, false, bb, (short)0, acc[t], false, false);
    }
    __syncthreads();
  }
  #pragma unroll
  for (int t = 0; t < 4; t++){
    int n = wn*64 + t*16 + (lane & 15);
    float bn = b2[n];
    #pragma unroll
    for (int v = 0; v < 8; v++){
      int m = m0 + wm*16 + v + ((lane < 16) ? 0 : 8);
      OUT[(size_t)m*128 + n] = acc[t][v] + bn;
    }
  }
}

// ===========================================================================
// Masked mean over time
// ===========================================================================
__global__ __launch_bounds__(128) void gc_kernel(
    const float* __restrict__ X2, const int* __restrict__ lens,
    float* __restrict__ GC)
{
  int b = blockIdx.x, d = threadIdx.x;
  int len = lens[b];
  float s = 0.f;
  for (int t = 0; t < len; t++) s += X2[((size_t)b*512 + t)*128 + d];
  GC[b*128 + d] = s / fmaxf((float)len, 1.f);
}

// ===========================================================================
// Pooling attention weights: FI[b,t] = mean_h softmax_t(q_h . k_h(t)/sqrt(32))
// ===========================================================================
__global__ __launch_bounds__(128) void pool_kernel(
    const float* __restrict__ GC, const float* __restrict__ X2,
    const float* __restrict__ Wqkv, const float* __restrict__ bqkv,
    const int* __restrict__ lens, float* __restrict__ FI)
{
  __shared__ float qs[128];
  __shared__ float st[4][512];
  int b = blockIdx.x, tid = threadIdx.x, lane = tid & 31, w = tid >> 5;
  int len = lens[b];
  {
    float a = bqkv[tid];
    const float4* gr = (const float4*)&GC[b*128];
    const float4* wr = (const float4*)&Wqkv[(size_t)tid*128];
    for (int c = 0; c < 32; c++){
      float4 x4 = gr[c], w4 = wr[c];
      a += x4.x*w4.x + x4.y*w4.y + x4.z*w4.z + x4.w*w4.w;
    }
    qs[tid] = a;
  }
  __syncthreads();
  const float scale = 0.17677669529663687f;
  for (int t = tid; t < 512; t += 128){
    const float4* xr = (const float4*)&X2[((size_t)b*512 + t)*128];
    float sc[4] = {0.f, 0.f, 0.f, 0.f};
    for (int o = 0; o < 128; o++){
      float kv = bqkv[128 + o];
      const float4* wr = (const float4*)&Wqkv[(size_t)(128+o)*128];
      for (int c = 0; c < 32; c++){
        float4 x4 = xr[c], w4 = wr[c];
        kv += x4.x*w4.x + x4.y*w4.y + x4.z*w4.z + x4.w*w4.w;
      }
      sc[o >> 5] += qs[o] * kv;
    }
    bool mask = (t >= len);
    #pragma unroll
    for (int h = 0; h < 4; h++) st[h][t] = mask ? -1e9f : sc[h]*scale;
  }
  __syncthreads();
  float mx = -3.4e38f;
  for (int t = lane; t < 512; t += 32) mx = fmaxf(mx, st[w][t]);
  mx = wred_max(mx);
  float sum = 0.f;
  for (int t = lane; t < 512; t += 32){
    float e = __expf(st[w][t] - mx);
    st[w][t] = e; sum += e;
  }
  sum = wred_sum(sum);
  float inv = 1.f / sum;
  for (int t = lane; t < 512; t += 32) st[w][t] *= inv;
  __syncthreads();
  for (int t = tid; t < 512; t += 128)
    FI[(size_t)b*512 + t] = 0.25f * (st[0][t] + st[1][t] + st[2][t] + st[3][t]);
}

// ===========================================================================
// Final: combined softmax pooling + classifier -> out (B,2)
// ===========================================================================
__global__ __launch_bounds__(128) void final_kernel(
    const float* __restrict__ X2, const float* __restrict__ FI,
    const float* __restrict__ projW, const float* __restrict__ projB,
    const int* __restrict__ lens,
    const float* __restrict__ W1, const float* __restrict__ b1,
    const float* __restrict__ bng, const float* __restrict__ bnb,
    const float* __restrict__ W2, const float* __restrict__ b2,
    float* __restrict__ out)
{
  __shared__ float cl[512];
  __shared__ __align__(16) float vr[128];
  __shared__ float hb[32];
  __shared__ float red[4];
  int b = blockIdx.x, tid = threadIdx.x, lane = tid & 31, w = tid >> 5;
  int len = lens[b];
  for (int t = tid; t < 512; t += 128){
    const float4* xr = (const float4*)&X2[((size_t)b*512 + t)*128];
    float a = projB[0];
    for (int d = 0; d < 32; d++){
      float4 x4 = xr[d], w4 = ((const float4*)projW)[d];
      a += x4.x*w4.x + x4.y*w4.y + x4.z*w4.z + x4.w*w4.w;
    }
    cl[t] = ((t < len) ? a : -1e9f) + FI[(size_t)b*512 + t]*10.f;
  }
  __syncthreads();
  float mx = -3.4e38f;
  for (int t = tid; t < 512; t += 128) mx = fmaxf(mx, cl[t]);
  mx = wred_max(mx);
  if (lane == 0) red[w] = mx;
  __syncthreads();
  mx = fmaxf(fmaxf(red[0], red[1]), fmaxf(red[2], red[3]));
  float sum = 0.f;
  for (int t = tid; t < 512; t += 128){
    float e = __expf(cl[t] - mx);
    cl[t] = e; sum += e;
  }
  sum = wred_sum(sum);
  __syncthreads();
  if (lane == 0) red[w] = sum;
  __syncthreads();
  float inv = 1.f / (red[0] + red[1] + red[2] + red[3]);
  for (int t = tid; t < 512; t += 128) cl[t] *= inv;
  __syncthreads();
  if (tid < 128){
    float s = 0.f;
    for (int t = 0; t < 512; t++) s += cl[t] * X2[((size_t)b*512 + t)*128 + tid];
    vr[tid] = s;
  }
  __syncthreads();
  if (tid < 32){
    float a = b1[tid];
    const float4* wr = (const float4*)&W1[(size_t)tid*128];
    const float4* vv = (const float4*)vr;
    for (int d = 0; d < 32; d++){
      float4 x4 = vv[d], w4 = wr[d];
      a += x4.x*w4.x + x4.y*w4.y + x4.z*w4.z + x4.w*w4.w;
    }
    a = a * (bng[tid]*BN_RSQ) + bnb[tid];
    hb[tid] = fmaxf(a, 0.f);
  }
  __syncthreads();
  if (tid < 2){
    float a = b2[tid];
    for (int d = 0; d < 32; d++) a += hb[d]*W2[tid*32 + d];
    out[b*2 + tid] = a;
  }
}

// ===========================================================================
// Host launcher
// ===========================================================================
extern "C" void kernel_launch(void* const* d_in, const int* in_sizes, int n_in,
                              void* d_out, int out_size, void* d_ws, size_t ws_size,
                              hipStream_t stream)
{
  (void)in_sizes; (void)n_in; (void)out_size; (void)ws_size;
  auto F = [&](int i) -> const float* { return (const float*)d_in[i]; };
  const int* lens = (const int*)d_in[6];

  const size_t BT = 32768;   // 64 * 512
  float* ws = (float*)d_ws;
  float* Rb   = ws;                        // BT*384  (region feats, later QKV)
  float* PACK = Rb + BT*384;               // BT*704  (later transformer acts)
  float* Gb   = PACK + BT*704;             // BT*64
  float* CRb  = Gb + BT*64;                // BT*64
  float* PFb  = CRb + BT*64;               // BT*128
  float* GCb  = PFb + BT*128;              // 64*128
  float* FIb  = GCb + 64*128;              // BT
  f16*   globWh = (f16*)(FIb + BT);        // 64*704
  f16*   qkvWh  = globWh + 64*704;         // 384*128
  f16*   woWh   = qkvWh + 384*128;         // 128*128
  f16*   w1Wh   = woWh + 128*128;          // 2048*128
  f16*   w2Wh   = w1Wh + 2048*128;         // 128*2048
  // aliases into PACK (lifetimes disjoint with glob GEMM):
  float* QKVb = Rb;                        // BT*384
  float* AOr  = PACK;                      // BT*128
  float* AO2  = PACK + BT*128;             // BT*128
  float* Xb   = PACK + BT*256;             // BT*128
  float* FFO  = PACK + BT*384;             // BT*128
  float* X2b  = PACK + BT*512;             // BT*128

  // 0) pre-convert GEMM weights to padded f16
  {
    size_t t1 = (size_t)64*704, t2 = (size_t)384*128, t3 = (size_t)128*128;
    size_t t4 = (size_t)2048*128, t5 = (size_t)128*2048;
    w2h_kernel<<<(unsigned)((t1+255)/256),256,0,stream>>>(F(59), globWh, 690, 704, t1);
    w2h_kernel<<<(unsigned)((t2+255)/256),256,0,stream>>>(F(65), qkvWh, 128, 128, t2);
    w2h_kernel<<<(unsigned)((t3+255)/256),256,0,stream>>>(F(67), woWh, 128, 128, t3);
    w2h_kernel<<<(unsigned)((t4+255)/256),256,0,stream>>>(F(71), w1Wh, 128, 128, t4);
    w2h_kernel<<<(unsigned)((t5+255)/256),256,0,stream>>>(F(73), w2Wh, 2048, 2048, t5);
  }

  // 1) region encoders
  region_kernel< 9><<<4096,256,0,stream>>>(F(0), F( 7),F( 8),F( 9),F(10),F(11),F(12),
                                           F(43),F(44),F(45),F(46),F(55),F(56), Rb, 0);
  region_kernel< 9><<<4096,256,0,stream>>>(F(1), F(13),F(14),F(15),F(16),F(17),F(18),
                                           F(43),F(44),F(45),F(46),F(55),F(56), Rb, 1);
  region_kernel<11><<<4096,256,0,stream>>>(F(2), F(19),F(20),F(21),F(22),F(23),F(24),
                                           F(43),F(44),F(45),F(46),F(55),F(56), Rb, 2);
  region_kernel<11><<<4096,256,0,stream>>>(F(3), F(25),F(26),F(27),F(28),F(29),F(30),
                                           F(43),F(44),F(45),F(46),F(55),F(56), Rb, 3);
  region_kernel< 9><<<4096,256,0,stream>>>(F(4), F(31),F(32),F(33),F(34),F(35),F(36),
                                           F(43),F(44),F(45),F(46),F(55),F(56), Rb, 4);
  region_kernel<20><<<4096,256,0,stream>>>(F(5), F(37),F(38),F(39),F(40),F(41),F(42),
                                           F(43),F(44),F(45),F(46),F(55),F(56), Rb, 5);

  // 2) global branch: pack (padded to 704) + WMMA GEMM with BN+ReLU epilogue
  {
    size_t total = BT*704;
    pack_kernel<<<(unsigned)((total + 255)/256),256,0,stream>>>(F(0),F(1),F(2),F(3),F(4),F(5), PACK, total);
    gemm_wmma<<<dim3(2,512),256,0,stream>>>(PACK,704, globWh, F(60), F(61), F(62),
                                            Gb,64, 64,704, 2);
  }

  // 3) cross attention over regions -> CR
  cross_kernel<<<4096,256,0,stream>>>(Rb, F(47),F(48),F(49),F(50),F(57),F(58), CRb);

  // 4) per-frame fuse + LN + positional encoding
  perframe_kernel<<<4096,256,0,stream>>>(CRb, Gb, F(63), F(64), PFb);

  // 5) temporal transformer
  gemm_wmma<<<dim3(12,512),256,0,stream>>>(PFb,128, qkvWh, F(66), nullptr,nullptr,
                                           QKVb,384, 384,128, 0);             // QKV proj
  attn_kernel<<<dim3(32,4,64),32,0,stream>>>(QKVb, lens, AOr);                // masked MHA
  gemm_wmma<<<dim3(4,512),256,0,stream>>>(AOr,128, woWh, F(68), nullptr,nullptr,
                                          AO2,128, 128,128, 0);               // Wo
  ln_add128<<<4096,256,0,stream>>>(PFb, AO2, F(69), F(70), Xb);               // t_ln1
  ff_kernel<<<512,256,0,stream>>>(Xb, w1Wh, F(72), w2Wh, F(74), FFO);         // FFN
  ln_add128<<<4096,256,0,stream>>>(Xb, FFO, F(75), F(76), X2b);               // t_ln2

  // 6) pooling + classifier
  gc_kernel<<<64,128,0,stream>>>(X2b, lens, GCb);
  pool_kernel<<<64,128,0,stream>>>(GCb, X2b, F(51), F(52), lens, FIb);
  final_kernel<<<64,128,0,stream>>>(X2b, FIb, F(77), F(78), lens,
                                    F(79), F(80), F(81), F(82), F(83), F(84),
                                    (float*)d_out);
}
